// GRU_4234837754276
// MI455X (gfx1250) — matrix operand order repaired and checked by reading, not compiled, not run on hardware
//
#include <hip/hip_runtime.h>
#include <hip/hip_fp16.h>

// ---------------------------------------------------------------------------
// BiGRU (2 layers, B=T=H=256) for MI455X / gfx1250, wave32 + WMMA.
//
// Algebraic cut: output depends only on h2[:, -1, :]; layer-1 backward at
// t=T-1 is the FIRST step of its reverse scan from h0=0 -> single GEMM +
// elementwise, no scan.  Layer-0 (both dirs) and layer-1 fwd run fully.
//
// Scan engine (per 16-row batch tile, 16 waves):
//   * h state double-buffered in LDS in the WMMA 16x16x32-f16 A-fragment
//     swizzle -> A loads are raw ds_load_b128 pairs.
//   * R,Z gate weight fragments preloaded into explicit register arrays
//     (128 VGPRs, resident across the whole scan - round-3 asm confirms no
//     spills).  N-gate fragments are streamed from L2 each step: an empty
//     `asm volatile("" : "+v"(off))` on a 32-BIT OFFSET makes the address
//     loop-variant (LICM can't hoist) while keeping pointer provenance so
//     the loads stay global_load_b128 (a pointer-valued opaque produced
//     generic flat_load_b128 with xcnt/dscnt coupling; volatile produced
//     serialized SCOPE_SYS flats - both worse).
//   * v_wmma_f32_16x16x32_f16, f32 accumulators (h bounded by tanh/sigmoid).
//   * Layer-0 input projection (K=4) via VALU on the fly.
//   * h1 streamed out already in A-fragment layout -> layer-1 input GEMM
//     needs no transpose.
//
// xp GEMM: 2 M-tiles x 6 N-tiles per wave (12 accs) -> 0.67KB of fragment
// traffic per WMMA (~2.1GB total) instead of 2KB/WMMA.
//
// Scratch (~170MB): h1 frags 64MB, xp1 f16 96MB, weights/rest ~3.5MB.
// ---------------------------------------------------------------------------

typedef __attribute__((ext_vector_type(16))) _Float16 v16h;
typedef __attribute__((ext_vector_type(8)))  float    v8f;
typedef unsigned int __attribute__((ext_vector_type(4))) v4u;

#define WMMA_F16(a, b, c) \
  __builtin_amdgcn_wmma_f32_16x16x32_f16(false, (a), false, (b), (short)0, (c), false, false)

__device__ __forceinline__ float sigm(float v) { return 1.f / (1.f + __expf(-v)); }

// ---------------------------------------------------------------------------
// f32 W[768][K] row-major -> f16 B-fragment swizzle:
//   out[(ntile*(K/32)+kfrag)*32 + lane][16]
// lane L -> n = ntile*16 + (L&15); element i -> k = kfrag*32 + 8*(L>>4) + (i&7) + 16*(i>>3).
// B[k][n] = W[n][k]  (gh = h @ W.T).
// ---------------------------------------------------------------------------
__global__ void prep_bsw(const float* __restrict__ W, _Float16* __restrict__ out, int K) {
  int id   = blockIdx.x * 256 + threadIdx.x;
  int lane = id & 31;
  int rest = id >> 5;
  int kfc  = K >> 5;
  int ntile = rest / kfc;
  int kfrag = rest % kfc;
  if (ntile >= 48) return;
  int n = ntile * 16 + (lane & 15);
  int kbase = kfrag * 32 + 8 * (lane >> 4);
  v16h v;
#pragma unroll
  for (int i = 0; i < 16; ++i) v[i] = (_Float16)W[(size_t)n * K + kbase + (i & 7) + 16 * (i >> 3)];
  *(v16h*)(out + ((size_t)rest * 32 + lane) * 16) = v;
}

// ---------------------------------------------------------------------------
// Layer-0 scan: grid (16 batch tiles, 2 dirs), 512 threads = 16 waves.
// ---------------------------------------------------------------------------
__global__ __launch_bounds__(512, 1) void l0_scan(
    const float* __restrict__ x, const float* __restrict__ Wih0,
    const float* __restrict__ bih0v, const float* __restrict__ bhh0v,
    const _Float16* __restrict__ whh_sw, _Float16* __restrict__ h1sw) {
  const int tid = threadIdx.x;
  const int L = tid & 31;
  const int w = tid >> 5;                // gate column block
  const int bx = blockIdx.x;
  const int d = blockIdx.y;
  const int b0 = bx * 16;
  const int hcol = w * 16 + (L & 15);
  const int half = L >> 4;

  __shared__ __align__(32) _Float16 hbuf[2][8][32][16];
  __shared__ float xbuf[16][4];

  // per-lane constants (r/z biases combined: they always appear summed)
  float wih[3][4], brz0, brz1, bihn, bhhn;
  {
    int r0 = d * 768 + hcol, r1 = r0 + 256, r2 = r0 + 512;
#pragma unroll
    for (int f = 0; f < 4; ++f) {
      wih[0][f] = Wih0[r0 * 4 + f];
      wih[1][f] = Wih0[r1 * 4 + f];
      wih[2][f] = Wih0[r2 * 4 + f];
    }
    brz0 = bih0v[r0] + bhh0v[r0];
    brz1 = bih0v[r1] + bhh0v[r1];
    bihn = bih0v[r2];
    bhhn = bhh0v[r2];
  }
  const _Float16* wbase = whh_sw + (size_t)d * (48 * 8 * 512);

  // R,Z fragments: resident in registers for the whole scan (16 x v16h)
  v16h bRreg[8], bZreg[8];
#pragma unroll
  for (int kf = 0; kf < 8; ++kf) {
    bRreg[kf] = *(const v16h*)(wbase + (size_t)(((0 * 16 + w) * 8 + kf) * 32 + L) * 16);
    bZreg[kf] = *(const v16h*)(wbase + (size_t)(((1 * 16 + w) * 8 + kf) * 32 + L) * 16);
  }
  // N fragments: streamed; opaque 32-bit offset defeats LICM but preserves
  // pointer provenance (-> global_load_b128, LOADcnt only).
  const _Float16* wbaseN = wbase + (size_t)((2 * 16 + w) * 8) * 512 + (size_t)L * 16;
  unsigned int offN = 0;

  float hreg[8];
#pragma unroll
  for (int e = 0; e < 8; ++e) hreg[e] = 0.f;

  { // zero hbuf[0]
    v4u z = {0, 0, 0, 0};
    *(v4u*)((char*)&hbuf[0][0][0][0] + tid * 16) = z;
  }
  const int kl = hcol & 31;
  const int kfw = hcol >> 5;
  const int iw = 2 * (((kl >> 1) & 3) + 4 * (kl >> 4)) + (kl & 1);
  const int lane2b = 16 * ((kl >> 3) & 1);

  int pp = 0;
  for (int s = 0; s < 256; ++s) {
    int t = d ? (255 - s) : s;
    asm volatile("" : "+v"(offN));     // block hoisting of N-frag loads
    const _Float16* wN = wbaseN + offN;
    if (tid < 64) {
      int row = tid >> 2, f = tid & 3;
      xbuf[row][f] = x[((size_t)(b0 + row) * 256 + t) * 4 + f];
    }
    __syncthreads();   // xbuf staged + hbuf[pp] complete

    v8f acc0 = {}, acc1 = {}, acc2 = {};
#pragma unroll
    for (int kf = 0; kf < 8; ++kf) {
      v16h a = *(const v16h*)&hbuf[pp][kf][L][0];
      acc0 = WMMA_F16(a, bRreg[kf], acc0);
      acc1 = WMMA_F16(a, bZreg[kf], acc1);
      v16h bN = *(const v16h*)(wN + (size_t)kf * 512);
      acc2 = WMMA_F16(a, bN, acc2);
    }

    int wr = pp ^ 1;
#pragma unroll
    for (int e = 0; e < 8; ++e) {
      int m = e + 8 * half;
      float xr = xbuf[m][0] * wih[0][0] + xbuf[m][1] * wih[0][1] +
                 xbuf[m][2] * wih[0][2] + xbuf[m][3] * wih[0][3];
      float xz = xbuf[m][0] * wih[1][0] + xbuf[m][1] * wih[1][1] +
                 xbuf[m][2] * wih[1][2] + xbuf[m][3] * wih[1][3];
      float xn = bihn + xbuf[m][0] * wih[2][0] + xbuf[m][1] * wih[2][1] +
                 xbuf[m][2] * wih[2][2] + xbuf[m][3] * wih[2][3];
      float r = sigm(xr + acc0[e] + brz0);
      float z = sigm(xz + acc1[e] + brz1);
      float n = tanhf(xn + r * (acc2[e] + bhhn));
      float hn = (1.f - z) * n + z * hreg[e];
      hreg[e] = hn;
      hbuf[wr][kfw][m + lane2b][iw] = (_Float16)hn;
    }
    __syncthreads();   // hbuf[wr] complete

    { // stream h tile (already A-swizzled) to h1sw
      int kf = tid >> 6, rem = tid & 63, ln = rem >> 1, c = rem & 1;
      size_t goff = (((size_t)(t * 16 + bx) * 16 + d * 8 + kf) * 32 + ln) * 16 + c * 8;
      *(v4u*)(h1sw + goff) = *(const v4u*)&hbuf[wr][kf][ln][c * 8];
    }
    pp = wr;
  }
}

// ---------------------------------------------------------------------------
// xp = h1 @ Wih1[dir].T + bih.  2 M-tiles x 6 N-tiles per wave.
// grid (nMtilePairs, 1), 256 threads = 8 waves; wave w -> ntiles w*6..w*6+5.
// Output f16 row-major [rows][768], row = (mtile - mtile_base)*16 + m.
// ---------------------------------------------------------------------------
__global__ __launch_bounds__(256, 1) void xp_gemm(
    const _Float16* __restrict__ h1sw, const _Float16* __restrict__ wih_sw,
    const float* __restrict__ bih, _Float16* __restrict__ out, int mtile_base) {
  int tid = threadIdx.x, L = tid & 31, w = tid >> 5;
  int mt0 = mtile_base + blockIdx.x * 2;
  const _Float16* a0 = h1sw + (size_t)mt0 * 16 * 512;
  const _Float16* a1 = a0 + 16 * 512;
  const _Float16* bb = wih_sw + (size_t)(w * 6) * 16 * 512;

  v8f acc[2][6];
#pragma unroll
  for (int i = 0; i < 2; ++i)
#pragma unroll
    for (int j = 0; j < 6; ++j) acc[i][j] = {};

#pragma unroll
  for (int kf = 0; kf < 16; ++kf) {
    size_t fo = (size_t)(kf * 32 + L) * 16;
    v16h A0 = *(const v16h*)(a0 + fo);
    v16h A1 = *(const v16h*)(a1 + fo);
#pragma unroll
    for (int j = 0; j < 6; ++j) {
      v16h Bj = *(const v16h*)(bb + (size_t)j * 16 * 512 + fo);
      acc[0][j] = WMMA_F16(A0, Bj, acc[0][j]);
      acc[1][j] = WMMA_F16(A1, Bj, acc[1][j]);
    }
  }

  int half = L >> 4;
  size_t row0 = (size_t)blockIdx.x * 32;
#pragma unroll
  for (int j = 0; j < 6; ++j) {
    int n = (w * 6 + j) * 16 + (L & 15);
    float bias = bih[n];
#pragma unroll
    for (int e = 0; e < 8; ++e) {
      int m = e + 8 * half;
      out[(row0 + m) * 768 + n]      = (_Float16)(acc[0][j][e] + bias);
      out[(row0 + 16 + m) * 768 + n] = (_Float16)(acc[1][j][e] + bias);
    }
  }
}

// ---------------------------------------------------------------------------
// Layer-1 forward scan; only t=T-1 hidden state persisted.
// ---------------------------------------------------------------------------
__global__ __launch_bounds__(512, 1) void l1_scan(
    const _Float16* __restrict__ xp, const float* __restrict__ bhh1v,
    const _Float16* __restrict__ whh_sw, float* __restrict__ hlast) {
  const int tid = threadIdx.x;
  const int L = tid & 31;
  const int w = tid >> 5;
  const int b0 = blockIdx.x * 16;
  const int hcol = w * 16 + (L & 15);
  const int half = L >> 4;

  __shared__ __align__(32) _Float16 hbuf[2][8][32][16];

  float bhh[3];
#pragma unroll
  for (int g = 0; g < 3; ++g) bhh[g] = bhh1v[g * 256 + hcol];

  v16h bRreg[8], bZreg[8];
#pragma unroll
  for (int kf = 0; kf < 8; ++kf) {
    bRreg[kf] = *(const v16h*)(whh_sw + (size_t)(((0 * 16 + w) * 8 + kf) * 32 + L) * 16);
    bZreg[kf] = *(const v16h*)(whh_sw + (size_t)(((1 * 16 + w) * 8 + kf) * 32 + L) * 16);
  }
  const _Float16* wbaseN = whh_sw + (size_t)((2 * 16 + w) * 8) * 512 + (size_t)L * 16;
  unsigned int offN = 0;

  float hreg[8];
#pragma unroll
  for (int e = 0; e < 8; ++e) hreg[e] = 0.f;

  {
    v4u z = {0, 0, 0, 0};
    *(v4u*)((char*)&hbuf[0][0][0][0] + tid * 16) = z;
  }
  const int kl = hcol & 31;
  const int kfw = hcol >> 5;
  const int iw = 2 * (((kl >> 1) & 3) + 4 * (kl >> 4)) + (kl & 1);
  const int lane2b = 16 * ((kl >> 3) & 1);
  __syncthreads();

  int pp = 0;
  for (int t = 0; t < 256; ++t) {
    asm volatile("" : "+v"(offN));     // block hoisting of N-frag loads
    const _Float16* wN = wbaseN + offN;

    v8f acc0 = {}, acc1 = {}, acc2 = {};
#pragma unroll
    for (int kf = 0; kf < 8; ++kf) {
      v16h a = *(const v16h*)&hbuf[pp][kf][L][0];
      acc0 = WMMA_F16(a, bRreg[kf], acc0);
      acc1 = WMMA_F16(a, bZreg[kf], acc1);
      v16h bN = *(const v16h*)(wN + (size_t)kf * 512);
      acc2 = WMMA_F16(a, bN, acc2);
    }

    int wr = pp ^ 1;
#pragma unroll
    for (int e = 0; e < 8; ++e) {
      int m = e + 8 * half;
      size_t rbase = ((size_t)(t * 256 + b0 + m)) * 768 + hcol;
      float xr = (float)xp[rbase];
      float xz = (float)xp[rbase + 256];
      float xn = (float)xp[rbase + 512];
      float r = sigm(xr + acc0[e] + bhh[0]);
      float z = sigm(xz + acc1[e] + bhh[1]);
      float n = tanhf(xn + r * (acc2[e] + bhh[2]));
      float hn = (1.f - z) * n + z * hreg[e];
      hreg[e] = hn;
      hbuf[wr][kfw][m + lane2b][iw] = (_Float16)hn;
      if (t == 255) hlast[(b0 + m) * 256 + hcol] = hn;
    }
    __syncthreads();
    pp = wr;
  }
}

// Layer-1 backward at t=T-1 from h0=0: gh == bhh1[1] exactly.
__global__ void bwd_step(const _Float16* __restrict__ xpb,
                         const float* __restrict__ bhh, float* __restrict__ h2bwd) {
  int id = blockIdx.x * 256 + threadIdx.x;   // 65536
  int b = id >> 8, h = id & 255;
  float xr = (float)xpb[(size_t)b * 768 + h];
  float xz = (float)xpb[(size_t)b * 768 + 256 + h];
  float xn = (float)xpb[(size_t)b * 768 + 512 + h];
  float r = sigm(xr + bhh[h]);
  float z = sigm(xz + bhh[256 + h]);
  float n = tanhf(xn + r * bhh[512 + h]);
  h2bwd[id] = (1.f - z) * n;   // + z*0
}

// Output head: sum dirs, 3-logit GEMV, softmax.
__global__ void head(const float* __restrict__ hf, const float* __restrict__ hb,
                     const float* __restrict__ Wo, const float* __restrict__ bo,
                     float* __restrict__ out) {
  int b = threadIdx.x;
  float l0 = bo[0], l1 = bo[1], l2 = bo[2];
  for (int h = 0; h < 256; ++h) {
    float v = hf[b * 256 + h] + hb[b * 256 + h];
    l0 += v * Wo[h]; l1 += v * Wo[256 + h]; l2 += v * Wo[512 + h];
  }
  float mx = fmaxf(l0, fmaxf(l1, l2));
  float e0 = __expf(l0 - mx), e1 = __expf(l1 - mx), e2 = __expf(l2 - mx);
  float inv = 1.f / (e0 + e1 + e2);
  out[b * 3 + 0] = e0 * inv;
  out[b * 3 + 1] = e1 * inv;
  out[b * 3 + 2] = e2 * inv;
}

extern "C" void kernel_launch(void* const* d_in, const int* in_sizes, int n_in,
                              void* d_out, int out_size, void* d_ws, size_t ws_size,
                              hipStream_t stream) {
  (void)in_sizes; (void)n_in; (void)out_size; (void)ws_size;
  const float* x    = (const float*)d_in[0];
  const float* Wih0 = (const float*)d_in[1];
  const float* Whh0 = (const float*)d_in[2];
  const float* bih0 = (const float*)d_in[3];
  const float* bhh0 = (const float*)d_in[4];
  const float* Wih1 = (const float*)d_in[5];
  const float* Whh1 = (const float*)d_in[6];
  const float* bih1 = (const float*)d_in[7];
  const float* bhh1 = (const float*)d_in[8];
  const float* W_out = (const float*)d_in[9];
  const float* b_out = (const float*)d_in[10];

  char* ws = (char*)d_ws;
  size_t off = 0;
  auto take = [&](size_t bytes) -> char* {
    char* p = ws + off;
    off += (bytes + 255) & ~(size_t)255;
    return p;
  };
  _Float16* h1sw   = (_Float16*)take((size_t)4096 * 16 * 32 * 16 * 2); // 64MB, A-frag layout
  _Float16* xp1f   = (_Float16*)take((size_t)65536 * 768 * 2);         // 96MB
  _Float16* xpb    = (_Float16*)take((size_t)256 * 768 * 2);
  _Float16* whh0sw = (_Float16*)take((size_t)2 * 48 * 8 * 512 * 2);
  _Float16* whh1sw = (_Float16*)take((size_t)48 * 8 * 512 * 2);
  _Float16* wih1sw = (_Float16*)take((size_t)2 * 48 * 16 * 512 * 2);
  float* hflast    = (float*)take((size_t)65536 * 4);
  float* hbwd      = (float*)take((size_t)65536 * 4);

  // weight conversion / fragment swizzle (tiny, L2-resident afterwards)
  prep_bsw<<<48, 256, 0, stream>>>(Whh0,             whh0sw,                 256);
  prep_bsw<<<48, 256, 0, stream>>>(Whh0 + 768 * 256, whh0sw + 48 * 8 * 512,  256);
  prep_bsw<<<48, 256, 0, stream>>>(Whh1,             whh1sw,                 256);
  prep_bsw<<<96, 256, 0, stream>>>(Wih1,             wih1sw,                 512);
  prep_bsw<<<96, 256, 0, stream>>>(Wih1 + 768 * 512, wih1sw + 48 * 16 * 512, 512);

  // layer 0, both directions concurrently
  l0_scan<<<dim3(16, 2), 512, 0, stream>>>(x, Wih0, bih0, bhh0, whh0sw, h1sw);

  // layer-1 input projections: full for fwd; only t=T-1 rows for bwd
  xp_gemm<<<dim3(2048, 1), 256, 0, stream>>>(h1sw, wih1sw, bih1, xp1f, 0);
  xp_gemm<<<dim3(8, 1),    256, 0, stream>>>(h1sw, wih1sw + 48 * 16 * 512, bih1 + 768, xpb, 4080);

  // layer-1 forward scan; backward is a single closed-form step
  l1_scan<<<dim3(16, 1), 512, 0, stream>>>(xp1f, bhh1, whh1sw, hflast);
  bwd_step<<<256, 256, 0, stream>>>(xpb, bhh1 + 768, hbwd);

  head<<<1, 256, 0, stream>>>(hflast, hbwd, W_out, b_out, (float*)d_out);
}
